// GRUDemo_85298050498860
// MI455X (gfx1250) — matrix operand order repaired and checked by reading, not compiled, not run on hardware
//
#include <hip/hip_runtime.h>
#include <hip/hip_bf16.h>

// Persistent 2-layer GRU for MI455X (gfx1250, wave32, WMMA f32_16x16x32_f16).
// Each wave owns a 16-row batch tile through all 256 timesteps.
// S=256, B=8192, IN=1, H=32, 3H=96.
//
// v4 = v2: the v3 0.5-prescale experiment regressed the
// static instruction mix (regalloc/scheduling noise outweighed the saved op), so
// this reverts to unit-scale fragments with sig(x)=0.5*tanh(0.5x)+0.5.

#define S_LEN 256
#define BATCH 8192
#define HDIM  32
#define G3    96

typedef _Float16 f16;
typedef __attribute__((ext_vector_type(16))) _Float16 v16h;
typedef __attribute__((ext_vector_type(8)))  _Float16 v8h;
typedef __attribute__((ext_vector_type(8)))  float    v8f;
typedef __attribute__((ext_vector_type(4)))  float    v4f;

#define WMMA16x16x32(a, b, c) \
    __builtin_amdgcn_wmma_f32_16x16x32_f16(false, (a), false, (b), (short)0, (c), false, false)

__device__ __forceinline__ float fast_tanh(float x) {
#if __has_builtin(__builtin_amdgcn_tanhf)
    return __builtin_amdgcn_tanhf(x);           // V_TANH_F32 (TRANS32)
#else
    float e = __expf(-2.0f * __builtin_fabsf(x));
    float t = fmaf(-2.0f * e, __builtin_amdgcn_rcpf(1.0f + e), 1.0f);
    return x >= 0.0f ? t : -t;
#endif
}
__device__ __forceinline__ float fast_sig(float x) {
    return fmaf(0.5f, fast_tanh(0.5f * x), 0.5f);
}

// B-operand fragment (32x16 f16) for N-tile `nt` of a (96 x 32) row-major f32 weight
// matrix, interpreted as B = W^T (K=32, N=96). Contiguous K order (matches A fragments):
// lanes 0-15 hold K=0..15, lanes 16-31 hold K=16..31; VGPR j packs K=2j,2j+1; N = lane%16.
__device__ __forceinline__ v16h pack_bfrag(const float* __restrict__ W, int nt, int lane) {
    const int ng = nt * 16 + (lane & 15);   // gate column (N)
    const int kb = (lane >> 4) * 16;        // K half
    const float* r = W + ng * HDIM + kb;
    v16h f;
#pragma unroll
    for (int j = 0; j < 8; ++j) {
        f[2 * j]     = (f16)r[2 * j];
        f[2 * j + 1] = (f16)r[2 * j + 1];
    }
    return f;
}

// A-operand fragment (16x32 f16, M=batch rows, K=hidden) from a row-major LDS tile.
// Layout (matches B fragments): lanes 0-15 hold K0..15, lanes 16-31 hold K16..31.
__device__ __forceinline__ v16h load_afrag(const f16* tile, int lane) {
    const int row = lane & 15;
    const int kb  = (lane >> 4) * 16;
    v8h lo = *(const v8h*)(tile + row * HDIM + kb);
    v8h hh = *(const v8h*)(tile + row * HDIM + kb + 8);
    return __builtin_shufflevector(lo, hh, 0,1,2,3,4,5,6,7,8,9,10,11,12,13,14,15);
}

__global__ __launch_bounds__(32, 1) void gru2_persistent_kernel(
    const float* __restrict__ x,    const float* __restrict__ hin,
    const float* __restrict__ Wih0, const float* __restrict__ Whh0,
    const float* __restrict__ bih0, const float* __restrict__ bhh0,
    const float* __restrict__ Wih1, const float* __restrict__ Whh1,
    const float* __restrict__ bih1, const float* __restrict__ bhh1,
    const float* __restrict__ Wout, const float* __restrict__ bout,
    float* __restrict__ y, float* __restrict__ hstate)
{
    __shared__ float s_wih0[G3], s_bih0[G3], s_bhh0[G3], s_bih1[G3], s_bhh1[G3];
    __shared__ float s_wout[HDIM];
    __shared__ float s_bout;
    __shared__ __align__(64) f16 h0s[16][HDIM];   // layer0 hidden (== layer1 input), A layout
    __shared__ __align__(64) f16 h1s[16][HDIM];   // layer1 hidden, A layout
    // Per-lane pre-packed B fragments: [0..5]=W_hh0, [6..11]=W_ih1, [12..17]=W_hh1
    __shared__ __align__(32) f16 sWB[18 * 32 * 16];

    const int lane   = threadIdx.x;      // 32 threads = 1 wave
    const int hi     = lane >> 4;        // lane half
    const int col    = lane & 15;        // N within a 16-wide tile
    const int batch0 = blockIdx.x * 16;  // this wave's batch rows

    // ---- one-time setup: constants to LDS ----
    for (int g = lane; g < G3; g += 32) {
        s_wih0[g] = Wih0[g];
        s_bih0[g] = bih0[g];
        s_bhh0[g] = bhh0[g];
        s_bih1[g] = bih1[g];
        s_bhh1[g] = bhh1[g];
    }
    if (lane < HDIM) s_wout[lane] = Wout[lane];
    if (lane == 0)   s_bout = bout[0];

    // ---- pre-pack all WMMA B fragments per lane into LDS ----
#pragma unroll
    for (int nt = 0; nt < 6; ++nt) {
        *(v16h*)(sWB + ((nt      * 32 + lane) << 4)) = pack_bfrag(Whh0, nt, lane);
        *(v16h*)(sWB + (((6+nt)  * 32 + lane) << 4)) = pack_bfrag(Wih1, nt, lane);
        *(v16h*)(sWB + (((12+nt) * 32 + lane) << 4)) = pack_bfrag(Whh1, nt, lane);
    }

    // ---- initial hidden state: registers (C/D layout) + LDS (A layout) ----
    // C/D layout: VGPR v, lane -> row m = v + 8*hi, col j = col + 16*tt
    v8f h0cd[2], h1cd[2];
#pragma unroll
    for (int tt = 0; tt < 2; ++tt) {
#pragma unroll
        for (int v = 0; v < 8; ++v) {
            const int m = v + 8 * hi, j = col + 16 * tt;
            float a = hin[(batch0 + m) * HDIM + j];
            float b = hin[(size_t)BATCH * HDIM + (batch0 + m) * HDIM + j];
            h0cd[tt][v] = a;
            h1cd[tt][v] = b;
            h0s[m][j] = (f16)a;
            h1s[m][j] = (f16)b;
        }
    }
    __syncthreads();  // single-wave WG barrier + compiler fence

    // ---- hoist per-lane loop-invariant gate constants ----
    float c_wr[2], c_wz[2], c_wn[2];
    float c_br0[2], c_bz0[2], c_bin0[2], c_bhn0[2];
    float c_br1[2], c_bz1[2], c_bin1[2], c_bhn1[2];
#pragma unroll
    for (int tt = 0; tt < 2; ++tt) {
        const int jg = col + 16 * tt;
        c_wr[tt]   = s_wih0[jg];
        c_wz[tt]   = s_wih0[32 + jg];
        c_wn[tt]   = s_wih0[64 + jg];
        c_br0[tt]  = s_bih0[jg]      + s_bhh0[jg];
        c_bz0[tt]  = s_bih0[32 + jg] + s_bhh0[32 + jg];
        c_bin0[tt] = s_bih0[64 + jg];
        c_bhn0[tt] = s_bhh0[64 + jg];
        c_br1[tt]  = s_bih1[jg]      + s_bhh1[jg];
        c_bz1[tt]  = s_bih1[32 + jg] + s_bhh1[32 + jg];
        c_bin1[tt] = s_bih1[64 + jg];
        c_bhn1[tt] = s_bhh1[64 + jg];
    }
    const float r_bout = s_bout;

    // ================= time loop (serial recurrence) =================
    for (int t = 0; t < S_LEN; ++t) {
        // x tile: IN==1, so gi0 is an outer product folded into the C seed.
        const float* xp = x + (size_t)t * BATCH + batch0 + 8 * hi;
        v4f xa = *(const v4f*)(xp);
        v4f xb = *(const v4f*)(xp + 4);
        float xv[8] = {xa[0], xa[1], xa[2], xa[3], xb[0], xb[1], xb[2], xb[3]};

        // -------- layer 0 --------
        v16h a0 = load_afrag(&h0s[0][0], lane);  // old h0, before overwrite
#pragma unroll
        for (int tt = 0; tt < 2; ++tt) {
            const int jg = col + 16 * tt;
            const v16h w_r = *(const v16h*)(sWB + (((tt)     * 32 + lane) << 4));
            const v16h w_z = *(const v16h*)(sWB + (((2 + tt) * 32 + lane) << 4));
            const v16h w_n = *(const v16h*)(sWB + (((4 + tt) * 32 + lane) << 4));
            v8f cr, cz;
            const v8f zeroC = {};
            float gin[8];
#pragma unroll
            for (int v = 0; v < 8; ++v) {
                cr[v]  = fmaf(xv[v], c_wr[tt], c_br0[tt]);   // x*Wih + bih + bhh
                cz[v]  = fmaf(xv[v], c_wz[tt], c_bz0[tt]);
                gin[v] = fmaf(xv[v], c_wn[tt], c_bin0[tt]);  // gi_n (VALU, IN==1)
            }
            cr      = WMMA16x16x32(a0, w_r, cr);
            cz      = WMMA16x16x32(a0, w_z, cz);
            v8f chn = WMMA16x16x32(a0, w_n, zeroC);          // gh_n (bias folded below)
            v8f hn;
#pragma unroll
            for (int v = 0; v < 8; ++v) {
                float r = fast_sig(cr[v]);
                float z = fast_sig(cz[v]);
                float n = fast_tanh(fmaf(r, chn[v] + c_bhn0[tt], gin[v]));
                hn[v] = fmaf(z, h0cd[tt][v] - n, n);         // n + z*(h-n)
            }
            h0cd[tt] = hn;
#pragma unroll
            for (int v = 0; v < 8; ++v) h0s[v + 8 * hi][jg] = (f16)hn[v];
        }
        __syncthreads();  // order h0s stores before layer-1 A-frag loads

        // -------- layer 1 --------
        v16h a1 = load_afrag(&h0s[0][0], lane);  // new h0 = layer-1 input
        v16h ah = load_afrag(&h1s[0][0], lane);  // old h1
#pragma unroll
        for (int tt = 0; tt < 2; ++tt) {
            const int jg = col + 16 * tt;
            const v16h wir = *(const v16h*)(sWB + (((6 + tt)  * 32 + lane) << 4));
            const v16h wiz = *(const v16h*)(sWB + (((8 + tt)  * 32 + lane) << 4));
            const v16h win = *(const v16h*)(sWB + (((10 + tt) * 32 + lane) << 4));
            const v16h whr = *(const v16h*)(sWB + (((12 + tt) * 32 + lane) << 4));
            const v16h whz = *(const v16h*)(sWB + (((14 + tt) * 32 + lane) << 4));
            const v16h whn = *(const v16h*)(sWB + (((16 + tt) * 32 + lane) << 4));
            const v8f zeroC = {};
            v8f cr  = WMMA16x16x32(a1, wir, zeroC);
            cr      = WMMA16x16x32(ah, whr, cr);
            v8f cz  = WMMA16x16x32(a1, wiz, zeroC);
            cz      = WMMA16x16x32(ah, whz, cz);
            v8f cin = WMMA16x16x32(a1, win, zeroC);   // gi_n
            v8f chn = WMMA16x16x32(ah, whn, zeroC);   // gh_n
            v8f hn;
#pragma unroll
            for (int v = 0; v < 8; ++v) {
                float r = fast_sig(cr[v] + c_br1[tt]);
                float z = fast_sig(cz[v] + c_bz1[tt]);
                float n = fast_tanh(fmaf(r, chn[v] + c_bhn1[tt], cin[v] + c_bin1[tt]));
                hn[v] = fmaf(z, h1cd[tt][v] - n, n);
            }
            h1cd[tt] = hn;
#pragma unroll
            for (int v = 0; v < 8; ++v) h1s[v + 8 * hi][jg] = (f16)hn[v];
        }
        __syncthreads();  // order h1s stores before the output dot

        // -------- y_t = h1_new . W_out + b_out (lanes 0-15 own one row each) --------
        {
            const f16* hr = &h1s[0][0] + (size_t)(lane & 15) * HDIM;
            float acc = r_bout;
#pragma unroll
            for (int kk = 0; kk < 4; ++kk) {
                v8h hv = *(const v8h*)(hr + kk * 8);
#pragma unroll
                for (int i = 0; i < 8; ++i) acc = fmaf((float)hv[i], s_wout[kk * 8 + i], acc);
            }
            if (lane < 16) y[(size_t)t * BATCH + batch0 + (lane & 15)] = acc;
        }
    }

    // ---- final hidden state (L, B, H) from C/D-layout registers ----
#pragma unroll
    for (int tt = 0; tt < 2; ++tt) {
#pragma unroll
        for (int v = 0; v < 8; ++v) {
            const int m = v + 8 * hi, j = col + 16 * tt;
            hstate[(batch0 + m) * HDIM + j] = h0cd[tt][v];
            hstate[(size_t)BATCH * HDIM + (batch0 + m) * HDIM + j] = h1cd[tt][v];
        }
    }
}

extern "C" void kernel_launch(void* const* d_in, const int* in_sizes, int n_in,
                              void* d_out, int out_size, void* d_ws, size_t ws_size,
                              hipStream_t stream) {
    const float* x    = (const float*)d_in[0];
    const float* h    = (const float*)d_in[1];
    const float* Wih0 = (const float*)d_in[2];
    const float* Whh0 = (const float*)d_in[3];
    const float* bih0 = (const float*)d_in[4];
    const float* bhh0 = (const float*)d_in[5];
    const float* Wih1 = (const float*)d_in[6];
    const float* Whh1 = (const float*)d_in[7];
    const float* bih1 = (const float*)d_in[8];
    const float* bhh1 = (const float*)d_in[9];
    const float* Wout = (const float*)d_in[10];
    const float* bout = (const float*)d_in[11];
    float* y      = (float*)d_out;
    float* hstate = y + (size_t)S_LEN * BATCH;  // outputs concatenated flat

    dim3 grid(BATCH / 16);  // 512 single-wave workgroups (one 16-row batch tile each)
    dim3 block(32);
    gru2_persistent_kernel<<<grid, block, 0, stream>>>(
        x, h, Wih0, Whh0, bih0, bhh0, Wih1, Whh1, bih1, bhh1, Wout, bout, y, hstate);
}
